// GNNLayer_89979564851856
// MI455X (gfx1250) — matrix operand order repaired and checked by reading, not compiled, not run on hardware
//
#include <hip/hip_runtime.h>
#include <hip/hip_bf16.h>

typedef __attribute__((ext_vector_type(16))) _Float16     v16h;
typedef __attribute__((ext_vector_type(2)))  _Float16     v2h;
typedef __attribute__((ext_vector_type(8)))  float        v8f;
typedef __attribute__((ext_vector_type(4)))  unsigned int v4u;
typedef __attribute__((ext_vector_type(8)))  int          v8i;
typedef __attribute__((ext_vector_type(4)))  int          v4i;
typedef __attribute__((ext_vector_type(4)))  unsigned int u4;

#define D 128
#define SMAX 64
#define BFRAG_HALFS (8 * 4 * 32 * 16)   // full W1^T fragment image: 16384 halfs = 32 KB

// wscol[k] = sum_j w2[j][k]; wscol[D] = sum_j b2[j];
// w1f = f16 W1^T pre-swizzled into exact WMMA B-fragment order (shared by all blocks)
__global__ void prep_kernel(const float* __restrict__ w2, const float* __restrict__ b2,
                            const float* __restrict__ w1, float* __restrict__ wscol,
                            _Float16* __restrict__ w1f) {
  int t = threadIdx.x;
  if (t < D) {
    float s = 0.f;
    for (int j = 0; j < D; ++j) s += w2[j * D + t];
    wscol[t] = s;
  }
  if (t == 0) {
    float s = 0.f;
    for (int j = 0; j < D; ++j) s += b2[j];
    wscol[D] = s;
  }
  // B 32x16 packing used here (k order matches A): lane = g*16 + (n&15); half i: k = 16g + i
  for (int it = 0; it < 64; ++it) {
    int p  = it * 128 + t;
    int n  = p >> 6;            // output feature 0..127
    int q  = p & 63;
    int k0 = q << 1;
    float f0 = w1[n * D + k0];
    float f1 = w1[n * D + k0 + 1];
    int tt = n >> 4, lnr = n & 15;
    int kc = k0 >> 5, kin = k0 & 31;
    int gg = kin >> 4;
    int ii = kin & 15;                               // even; ii+1 holds k0+1
    int addr = (((tt * 4 + kc) * 32) + (gg * 16 + lnr)) * 16 + ii;
    *(v2h*)&w1f[addr] = (v2h){(_Float16)f0, (_Float16)f1};
  }
}

__global__ void __launch_bounds__(128)
head_kernel(const float* __restrict__ drug_emb, const float* __restrict__ rel_emb,
            const float* __restrict__ tail_emb, const _Float16* __restrict__ w1f,
            const float* __restrict__ b1, const float* __restrict__ fc2_w,
            const float* __restrict__ fc2_b, const int* __restrict__ head_idx,
            const int* __restrict__ rel_idx, const int* __restrict__ tail_idx,
            const int* __restrict__ sample_ptr, const float* __restrict__ wscol,
            float* __restrict__ out_pre, int n_edges) {
  __shared__ v16h s_af[4 * 4 * 32];   // A fragments [m-block][kc][lane]  16 KB
  __shared__ v16h s_bf[8 * 4 * 32];   // B fragments [n-tile][kc][lane]   32 KB
  __shared__ float s_drug[D], s_ws[D], s_b1[D];
  __shared__ float s_score[SMAX];
  __shared__ int   s_rel[SMAX], s_tail[SMAX];
  __shared__ float s_concat[2 * D];
  __shared__ int   s_range[2];

  _Float16* s_afh = (_Float16*)s_af;

  const int h    = blockIdx.x;
  const int tid  = threadIdx.x;
  const int lane = tid & 31;
  const int wave = tid >> 5;

  if (tid == 0) {  // lower_bound(h), lower_bound(h+1) in sorted head array
    int lo = 0, hi = n_edges;
    while (lo < hi) { int mid = (lo + hi) >> 1; if (head_idx[mid] < h) lo = mid + 1; else hi = mid; }
    s_range[0] = lo;
    hi = n_edges;
    while (lo < hi) { int mid = (lo + hi) >> 1; if (head_idx[mid] < h + 1) lo = mid + 1; else hi = mid; }
    s_range[1] = lo;
  }

  // ---- TDM: DMA the 32 KB pre-swizzled B-fragment image into LDS (wave 0) ----
#if __has_builtin(__builtin_amdgcn_tensor_load_to_lds)
  if (tid < 32) {
    unsigned lds_off = (unsigned)(uintptr_t)(void*)&s_bf[0];
    unsigned long long ga = (unsigned long long)(uintptr_t)w1f;
    v4u g0;
    g0[0] = 1u;                                   // count=1, user descriptor
    g0[1] = lds_off;                              // lds_addr (bytes)
    g0[2] = (unsigned)ga;                         // global_addr[31:0]
    g0[3] = (unsigned)(ga >> 32) | (2u << 30);    // global_addr[56:32] | type=2
    v8i g1;
    g1[0] = (3 << 16);                            // workgroup_mask=0, data_size=8B
    g1[1] = (int)(4096u << 16);                   // tensor_dim0[15:0]=4096
    g1[2] = (int)(1u << 16);                      // tensor_dim0[31:16]=0 | tensor_dim1=1
    g1[3] = (int)(4096u << 16);                   // tile_dim0=4096 (x 8B = 32 KB)
    g1[4] = 1;                                    // tile_dim1=1, tile_dim2=0
    g1[5] = 4096;                                 // tensor_dim0_stride
    g1[6] = 0; g1[7] = 0;
    v4i gz = {0, 0, 0, 0};
#if __clang_major__ >= 23
    v8i gz8 = {0, 0, 0, 0, 0, 0, 0, 0};
    __builtin_amdgcn_tensor_load_to_lds(g0, g1, gz, gz, gz8, 0);
#else
    __builtin_amdgcn_tensor_load_to_lds(g0, g1, gz, gz, 0);
#endif
    __builtin_amdgcn_s_wait_tensorcnt(0);
  }
#else
  {
    const u4* src = (const u4*)w1f;
    u4* dst = (u4*)s_bf;
    for (int i = tid; i < (int)(BFRAG_HALFS * 2 / 16); i += 128) dst[i] = src[i];
  }
#endif

  s_drug[tid] = drug_emb[h * D + tid];
  s_ws[tid]   = wscol[tid];
  s_b1[tid]   = b1[tid];
  __syncthreads();

  const int start = s_range[0];
  int smp = sample_ptr[0]; if (smp > SMAX) smp = SMAX;
  int nk = s_range[1] - start;
  if (nk > smp) nk = smp; if (nk < 0) nk = 0;

  if (tid < SMAX) {
    bool v = tid < nk;
    s_rel[tid]  = v ? rel_idx[start + tid]  : 0;
    s_tail[tid] = v ? tail_idx[start + tid] : 0;
  }
  __syncthreads();

  // Prefetch the scattered tail rows; consumed after the GEMM phase.
  if (tid < nk) __builtin_prefetch(&tail_emb[(long)s_tail[tid] * D], 0, 0);

  // ---- Stage A = HAD[64,128] (f16) directly in WMMA A-fragment order ----
  // A 16x32 packing used here (k order matches B): lane = g*16 + (m&15); half i: k = 16g + i
  #pragma unroll 4
  for (int it = 0; it < 32; ++it) {
    int p  = it * 128 + tid;
    int e  = p >> 6;            // edge row 0..63
    int q  = p & 63;            // half-pair; k = 2q, 2q+1
    int k0 = q << 1;
    float f0 = 0.f, f1 = 0.f;
    if (e < nk) {
      const float* rrow = rel_emb + (long)s_rel[e] * D;
      f0 = s_drug[k0]     * rrow[k0];
      f1 = s_drug[k0 + 1] * rrow[k0 + 1];
    }
    int wv = e >> 4, lnr = e & 15;
    int kc = k0 >> 5, kin = k0 & 31;
    int gg = kin >> 4;
    int ii = kin & 15;                               // even; ii+1 = k0+1
    int addr = (((wv * 4 + kc) * 32) + (gg * 16 + lnr)) * 16 + ii;
    *(v2h*)&s_afh[addr] = (v2h){(_Float16)f0, (_Float16)f1};
  }
  __syncthreads();

  // ---- GEMM: H1[64,128] = HAD @ W1^T ; wave owns a 16-edge row block ----
  const int m0 = wave * 16;
  const int g  = lane >> 4;
  const int ln = lane & 15;

  v16h afrag[4];
  #pragma unroll
  for (int kc = 0; kc < 4; ++kc)
    afrag[kc] = s_af[(wave * 4 + kc) * 32 + lane];

  float psum[8];
  #pragma unroll
  for (int v = 0; v < 8; ++v) psum[v] = 0.f;

  #pragma unroll 2
  for (int t = 0; t < 8; ++t) {
    v8f acc = {};
    #pragma unroll
    for (int kc = 0; kc < 4; ++kc)
      acc = __builtin_amdgcn_wmma_f32_16x16x32_f16(false, afrag[kc], false,
                s_bf[(t * 4 + kc) * 32 + lane], (short)0, acc, false, false);
    // fused epilogue: sigmoid + dot with w2 column sums
    const float b1n = s_b1[t * 16 + ln];
    const float wsn = s_ws[t * 16 + ln];
    #pragma unroll
    for (int v = 0; v < 8; ++v) {
      float z  = acc[v] + b1n;
      float hv = 1.0f / (1.0f + __expf(-z));
      psum[v] += hv * wsn;
    }
  }
  // reduce the 16 lanes that share each output row (wave32, 16-lane halves)
  #pragma unroll
  for (int v = 0; v < 8; ++v) {
    float p = psum[v];
    p += __shfl_xor(p, 1, 32);
    p += __shfl_xor(p, 2, 32);
    p += __shfl_xor(p, 4, 32);
    p += __shfl_xor(p, 8, 32);
    psum[v] = p;
  }
  const float bsum = wscol[D];
  if (ln == 0) {
    #pragma unroll
    for (int v = 0; v < 8; ++v)
      s_score[m0 + g * 8 + v] = psum[v] + bsum;      // C layout: M = v + 8*g
  }
  __syncthreads();
  if (tid < SMAX && tid >= nk) s_score[tid] = 0.f;
  __syncthreads();

  // neigh[h][k] = sum_e score[e] * tail_emb[t_e][k]
  float aggr = 0.f;
  for (int e = 0; e < nk; ++e)
    aggr += s_score[e] * tail_emb[(long)s_tail[e] * D + tid];
  s_concat[tid]     = s_drug[tid];
  s_concat[D + tid] = aggr;
  __syncthreads();

  // fc2: out_pre[h][j] = concat . fc2_w[j,:] + fc2_b[j]   (tiny; scalar VALU)
  float o = fc2_b[tid];
  #pragma unroll 8
  for (int k = 0; k < 2 * D; ++k)
    o += s_concat[k] * fc2_w[tid * (2 * D) + k];
  out_pre[h * D + tid] = o;
}

__global__ void __launch_bounds__(256)
bn_kernel(const float* __restrict__ pre, const float* __restrict__ gamma,
          const float* __restrict__ beta, const float* __restrict__ x,
          float* __restrict__ out, int n_rows) {
  __shared__ float r1[256], r2[256];
  const int j = blockIdx.x;
  const int tid = threadIdx.x;
  float s1 = 0.f, s2 = 0.f;
  for (int i = tid; i < n_rows; i += 256) {
    float v = pre[i * D + j];
    s1 += v; s2 += v * v;
  }
  r1[tid] = s1; r2[tid] = s2;
  __syncthreads();
  for (int off = 128; off > 0; off >>= 1) {
    if (tid < off) { r1[tid] += r1[tid + off]; r2[tid] += r2[tid + off]; }
    __syncthreads();
  }
  const float inv_n = 1.0f / (float)n_rows;
  const float mean  = r1[0] * inv_n;
  const float var   = r2[0] * inv_n - mean * mean;   // population variance (jnp.var)
  const float scale = gamma[j] * rsqrtf(var + 1e-5f);
  const float shift = beta[j] - mean * scale;
  float* xout = out + (size_t)n_rows * D;
  for (int i = tid; i < n_rows; i += 256) {
    out[i * D + j]  = pre[i * D + j] * scale + shift;
    xout[i * D + j] = x[i * D + j];
  }
}

extern "C" void kernel_launch(void* const* d_in, const int* in_sizes, int n_in,
                              void* d_out, int out_size, void* d_ws, size_t ws_size,
                              hipStream_t stream) {
  const float* drug = (const float*)d_in[0];
  const float* rel  = (const float*)d_in[1];
  const float* tail = (const float*)d_in[2];
  const float* w1   = (const float*)d_in[3];
  const float* b1   = (const float*)d_in[4];
  const float* w2   = (const float*)d_in[5];
  const float* b2   = (const float*)d_in[6];
  const float* fc2w = (const float*)d_in[7];
  const float* fc2b = (const float*)d_in[8];
  const float* gm   = (const float*)d_in[9];
  const float* bt   = (const float*)d_in[10];
  const float* x    = (const float*)d_in[11];
  const int* hidx   = (const int*)d_in[12];
  const int* ridx   = (const int*)d_in[13];
  const int* tidx   = (const int*)d_in[14];
  const int* smp    = (const int*)d_in[15];

  const int n_drug  = in_sizes[0] / D;
  const int n_edges = in_sizes[12];

  float* wsf        = (float*)d_ws;
  float* wscol      = wsf;                       // 256 floats reserved
  _Float16* w1f     = (_Float16*)(wsf + 256);    // 16384 halfs = 32 KB (8192 floats)
  float* pre        = wsf + 256 + 8192;          // n_drug * D floats

  prep_kernel<<<1, 128, 0, stream>>>(w2, b2, w1, wscol, w1f);
  head_kernel<<<n_drug, 128, 0, stream>>>(drug, rel, tail, w1f, b1, fc2w, fc2b,
                                          hidx, ridx, tidx, smp, wscol, pre, n_edges);
  bn_kernel<<<D, 256, 0, stream>>>(pre, gm, bt, x, (float*)d_out, n_drug);
}
